// BaumWelch_58394375356877
// MI455X (gfx1250) — matrix operand order repaired and checked by reading, not compiled, not run on hardware
//
#include <hip/hip_runtime.h>

// Baum-Welch forward-backward, N=256 seqs, T=256 steps, K=32 states.
// One wave32 per sequence; lane j <-> state j. Per-step 1x32 @ 32x32 matvec
// done with two V_WMMA_F32_16X16X32_F16 (K=32 matches the HW shape exactly).
// exp(log_trans) tile staged in padded LDS (stride 33 -> conflict-free
// transposed reads) and reused for the eta outer-product store in bwd.
// Serial-scan critical path: fast v_rcp instead of IEEE div, A-matrix built
// from LDS b128 broadcast reads, and trans tiles prefetched 8 steps ahead
// (global_prefetch_b8) to hide HBM latency at 1-wave/SIMD occupancy.

#define NN 256
#define TT 256
#define KK 32
#define PF 8          // tiles prefetched ahead (8 * 4KB = 32KB in flight)

typedef __attribute__((ext_vector_type(16))) _Float16 v16h;
typedef __attribute__((ext_vector_type(8)))  float    v8f;

static __device__ __forceinline__ float fast_rcp(float x) {
#if __has_builtin(__builtin_amdgcn_rcpf)
    return __builtin_amdgcn_rcpf(x);
#else
    return 1.0f / x;
#endif
}

static __device__ __forceinline__ float wave_sum(float v) {
    #pragma unroll
    for (int off = 16; off; off >>= 1) v += __shfl_xor(v, off, 32);
    return v;
}

// Prefetch one 4KB tile: 32 lanes x 128B stride covers all 32 cachelines.
static __device__ __forceinline__ void prefetch_tile(const float* tp, int lane) {
    __builtin_prefetch(tp + lane * 32, 0, 3);
}

// Stage exp(32x32 f32 tile) into padded LDS [32][33]; coalesced float4 loads.
static __device__ __forceinline__ void stage_tile(const float* __restrict__ tp,
                                                  float* tile, int lane) {
    #pragma unroll
    for (int i = 0; i < 8; ++i) {
        int f = i * 128 + lane * 4;
        float4 v = *(const float4*)(tp + f);
        int r = f >> 5, c = f & 31;
        float* d = tile + r * 33 + c;
        d[0] = __expf(v.x); d[1] = __expf(v.y);
        d[2] = __expf(v.z); d[3] = __expf(v.w);
    }
}

// Build 16x32 f16 A-matrix whose row 0 (M=0) is the 32-wide vector staged in
// LDS `vec`; rows 1..15 zero. Layout per CDNA5 ISA 7.12.2:
//   lane 0  holds K=0..7 (v0-3) and K=16..23 (v4-7)
//   lane 16 holds K=8..15 (v0-3) and K=24..31 (v4-7)
// 4 ds_load_b128 broadcast reads replace 16 wait-serialized ds_bpermute.
static __device__ __forceinline__ v16h build_A(const float* vec, int lane) {
    const float4* v4 = (const float4*)vec;
    int b = (lane & 16) ? 2 : 0;
    float4 q0 = v4[b],     q1 = v4[b + 1];
    float4 q2 = v4[b + 4], q3 = v4[b + 5];
    v16h A;
    A[0]  = (_Float16)q0.x; A[1]  = (_Float16)q0.y;
    A[2]  = (_Float16)q0.z; A[3]  = (_Float16)q0.w;
    A[4]  = (_Float16)q1.x; A[5]  = (_Float16)q1.y;
    A[6]  = (_Float16)q1.z; A[7]  = (_Float16)q1.w;
    A[8]  = (_Float16)q2.x; A[9]  = (_Float16)q2.y;
    A[10] = (_Float16)q2.z; A[11] = (_Float16)q2.w;
    A[12] = (_Float16)q3.x; A[13] = (_Float16)q3.y;
    A[14] = (_Float16)q3.z; A[15] = (_Float16)q3.w;
    if (lane != 0 && lane != 16) {
        #pragma unroll
        for (int i = 0; i < 16; ++i) A[i] = (_Float16)0.0f;
    }
    return A;
}

// r[j] = sum_k vec[k] * M[k][j], where M = tile (transposed=false)
//                               or    M[k][j] = tile[j][k] (transposed=true).
// Two WMMAs cover output columns 0..15 and 16..31; result row M=0 lives in
// C-VGPR0 of lanes 0..15.
static __device__ __forceinline__ float wmma_matvec(const float* tile,
                                                    const float* vec,
                                                    int lane, bool transposed) {
    v16h A = build_A(vec, lane);
    int nn = lane & 15;
    int kb = (lane >> 1) & 8;   // B layout: lanes 0-15 hold K=0..7,16..23; 16-31 hold K=8..15,24..31
    v16h B0, B1;
    if (!transposed) {
        #pragma unroll
        for (int i = 0; i < 16; ++i) {
            B0[i] = (_Float16)tile[(kb + i + (i & 8)) * 33 + nn];
            B1[i] = (_Float16)tile[(kb + i + (i & 8)) * 33 + 16 + nn];
        }
    } else {
        #pragma unroll
        for (int i = 0; i < 16; ++i) {
            B0[i] = (_Float16)tile[nn * 33 + (kb + i + (i & 8))];
            B1[i] = (_Float16)tile[(16 + nn) * 33 + (kb + i + (i & 8))];
        }
    }
    v8f c = {};
    v8f d0 = __builtin_amdgcn_wmma_f32_16x16x32_f16(false, A, false, B0,
                                                    (short)0, c, false, false);
    v8f d1 = __builtin_amdgcn_wmma_f32_16x16x32_f16(false, A, false, B1,
                                                    (short)0, c, false, false);
    float r0 = __shfl(d0[0], lane & 15, 32);
    float r1 = __shfl(d1[0], lane & 15, 32);
    return (lane < 16) ? r0 : r1;
}

__global__ __launch_bounds__(32) void fwd_kernel(
    const float* __restrict__ log_emiss,   // [N,T,K]
    const float* __restrict__ log_trans,   // [N,T-1,K,K]
    const float* __restrict__ log_prior,   // [K]
    float* __restrict__ alpha,             // ws [N,T,K]
    float* __restrict__ scale)             // ws [N,T]
{
    __shared__ __align__(16) float vec[KK];
    __shared__ float tile[KK * 33];
    const int n = blockIdx.x;
    const int lane = threadIdx.x;
    const float* em = log_emiss + (size_t)n * TT * KK;
    const float* tr = log_trans + (size_t)n * (TT - 1) * KK * KK;
    float* al = alpha + (size_t)n * TT * KK;
    float* sc = scale + (size_t)n * TT;

    #pragma unroll
    for (int i = 0; i < PF; ++i)
        prefetch_tile(tr + (size_t)i * KK * KK, lane);

    // t = 0: a0 = exp(prior) * exp(emiss[0]), safe-normalized
    float a = __expf(log_prior[lane]) * __expf(em[lane]);
    float s = wave_sum(a);
    s = (s == 0.0f) ? 1.0f : s;
    a *= fast_rcp(s);
    al[lane] = a;
    if (lane == 0) sc[0] = s;

    for (int t = 1; t < TT; ++t) {
        vec[lane] = a;
        if (t - 1 + PF < TT - 1)
            prefetch_tile(tr + (size_t)(t - 1 + PF) * KK * KK, lane);
        stage_tile(tr + (size_t)(t - 1) * KK * KK, tile, lane);
        __syncthreads();
        float raw = wmma_matvec(tile, vec, lane, false);   // (a_prev @ T)_j
        float an = raw * __expf(em[t * KK + lane]);
        float sn = wave_sum(an);
        sn = (sn == 0.0f) ? 1.0f : sn;
        an *= fast_rcp(sn);
        al[t * KK + lane] = an;
        if (lane == 0) sc[t] = sn;
        a = an;
        __syncthreads();   // before tile/vec overwrite next iteration
    }
}

__global__ __launch_bounds__(32) void bwd_kernel(
    const int*   __restrict__ inp,         // [N,T]
    const float* __restrict__ log_emiss,   // [N,T,K]
    const float* __restrict__ log_trans,   // [N,T-1,K,K]
    const float* __restrict__ alpha,       // ws [N,T,K]
    const float* __restrict__ scale,       // ws [N,T]
    float* __restrict__ eta,               // out [N,T,K,K]
    float* __restrict__ gamma)             // out [N,T,K]
{
    __shared__ __align__(16) float vec[KK];   // holds bmsg (buf)
    __shared__ float tile[KK * 33];
    const int n = blockIdx.x;
    const int lane = threadIdx.x;
    const int*   in = inp + (size_t)n * TT;
    const float* em = log_emiss + (size_t)n * TT * KK;
    const float* tr = log_trans + (size_t)n * (TT - 1) * KK * KK;
    const float* al = alpha + (size_t)n * TT * KK;
    const float* sc = scale + (size_t)n * TT;
    float* et = eta + (size_t)n * TT * KK * KK;
    float* gm = gamma + (size_t)n * TT * KK;

    #pragma unroll
    for (int i = 0; i < PF; ++i)
        prefetch_tile(tr + (size_t)(TT - 2 - i) * KK * KK, lane);

    // t = T-1: beta_last = 1/scale[T-1]; gamma[T-1]; eta[T-1] = 0
    float beta = fast_rcp(sc[TT - 1]);
    bool mask_next = (in[TT - 1] != 0);
    {
        float at = al[(TT - 1) * KK + lane];
        float g = at * beta;
        float s = wave_sum(g);
        s = (s == 0.0f) ? 1.0f : s;
        gm[(TT - 1) * KK + lane] = g * fast_rcp(s) * (mask_next ? 1.0f : 0.0f);
        float4 z4 = make_float4(0.0f, 0.0f, 0.0f, 0.0f);
        float4* ez = (float4*)(et + (size_t)(TT - 1) * KK * KK);
        #pragma unroll
        for (int i = 0; i < 8; ++i) ez[i * 32 + lane] = z4;
    }

    for (int t = TT - 2; t >= 0; --t) {
        float buf = beta * __expf(em[(t + 1) * KK + lane]);  // bmsg_j
        vec[lane] = buf;
        if (t - PF >= 0)
            prefetch_tile(tr + (size_t)(t - PF) * KK * KK, lane);
        stage_tile(tr + (size_t)t * KK * KK, tile, lane);
        __syncthreads();
        // b_raw[j] = sum_k bmsg[k] * T[j][k]   (transposed matvec)
        float braw = wmma_matvec(tile, vec, lane, true);
        float at = al[t * KK + lane];
        // eta normalizer z = sum_k alpha[k] * b_raw[k] (free from the matvec)
        float z = wave_sum(at * braw);
        bool mask_t = (in[t] != 0);
        float izm = fast_rcp(z) * (mask_next ? 1.0f : 0.0f);
        // eta[t][r][c] = alpha[r] * bmsg[c] * ptran[r][c] * izm, b128 stores
        float* erow = et + (size_t)t * KK * KK;
        const float4* bv4 = (const float4*)vec;
        #pragma unroll
        for (int i = 0; i < 8; ++i) {
            int f = i * 128 + lane * 4;
            int r = f >> 5, c = f & 31;
            float ak = __shfl(at, r, 32) * izm;
            float4 bq = bv4[c >> 2];
            const float* tl = tile + r * 33 + c;
            float4 o;
            o.x = ak * bq.x * tl[0];
            o.y = ak * bq.y * tl[1];
            o.z = ak * bq.z * tl[2];
            o.w = ak * bq.w * tl[3];
            *(float4*)(erow + f) = o;
        }
        // beta[t] = where(eos, 1, b_raw) / scale[t]
        bool eos = (mask_t != mask_next);
        beta = (eos ? 1.0f : braw) * fast_rcp(sc[t]);
        // gamma[t] = safe_norm(alpha*beta) * mask
        float g = at * beta;
        float s = wave_sum(g);
        s = (s == 0.0f) ? 1.0f : s;
        gm[t * KK + lane] = g * fast_rcp(s) * (mask_t ? 1.0f : 0.0f);
        mask_next = mask_t;
        __syncthreads();
    }
}

__global__ __launch_bounds__(256) void reduce_kernel(
    const int*   __restrict__ inp,     // [N,T]
    const float* __restrict__ scale,   // ws [N,T]
    const float* __restrict__ gamma,   // out gamma region
    float* __restrict__ prior,         // out [K]
    float* __restrict__ loglik)        // out scalar
{
    __shared__ float lls[256];
    __shared__ float cns[256];
    const int tid = threadIdx.x;
    if (tid < KK) {
        float p = 0.0f;
        for (int n = 0; n < NN; ++n) p += gamma[(size_t)n * TT * KK + tid];
        prior[tid] = p;
    }
    float ll = 0.0f, cn = 0.0f;
    for (int i = tid; i < NN * TT; i += 256) {
        if (inp[i] != 0) { ll += logf(scale[i]); cn += 1.0f; }
    }
    lls[tid] = ll; cns[tid] = cn;
    __syncthreads();
    for (int off = 128; off; off >>= 1) {
        if (tid < off) { lls[tid] += lls[tid + off]; cns[tid] += cns[tid + off]; }
        __syncthreads();
    }
    if (tid == 0) *loglik = lls[0] / cns[0];
}

extern "C" void kernel_launch(void* const* d_in, const int* in_sizes, int n_in,
                              void* d_out, int out_size, void* d_ws, size_t ws_size,
                              hipStream_t stream) {
    const int*   inp       = (const int*)d_in[0];
    const float* log_emiss = (const float*)d_in[1];
    const float* log_trans = (const float*)d_in[2];
    const float* log_prior = (const float*)d_in[3];
    float* out = (float*)d_out;

    // workspace: alpha [N,T,K] + scale [N,T]  (~8.7 MB)
    float* alpha = (float*)d_ws;
    float* scale = alpha + (size_t)NN * TT * KK;

    // output layout: prior[32] | eta[N*T*K*K] | gamma[N*T*K] | loglik[1]
    float* prior  = out;
    float* eta    = out + KK;
    float* gamma  = eta + (size_t)NN * TT * KK * KK;
    float* loglik = gamma + (size_t)NN * TT * KK;

    fwd_kernel<<<NN, 32, 0, stream>>>(log_emiss, log_trans, log_prior, alpha, scale);
    bwd_kernel<<<NN, 32, 0, stream>>>(inp, log_emiss, log_trans, alpha, scale, eta, gamma);
    reduce_kernel<<<1, 256, 0, stream>>>(inp, scale, gamma, prior, loglik);
}
